// ThermoSNN_4776003633776
// MI455X (gfx1250) — matrix operand order repaired and checked by reading, not compiled, not run on hardware
//
#include <hip/hip_runtime.h>
#include <cstdint>
#include <cstddef>

#define TT 12
#define BB 4
#define CHN 16
#define HH 256
#define WWI 256
#define HWIMG 65536
#define TBN 48
#define NPIXT 3145728u     /* TBN*HWIMG */
#define NELEMT 50331648u   /* NPIXT*CHN */

typedef __attribute__((ext_vector_type(16))) __bf16 v16bf;
typedef __attribute__((ext_vector_type(8)))  float  v8f;
typedef __attribute__((ext_vector_type(4)))  float  f4;
typedef __attribute__((ext_vector_type(4)))  unsigned int u4;

union AU { v16bf v; u4 q[2]; };

/* ---- CDNA5 async global->LDS path (guarded; falls back to sync copy) ----- */
#if defined(__has_builtin)
#  if __has_builtin(__builtin_amdgcn_global_load_async_to_lds_b128) && \
      __has_builtin(__builtin_amdgcn_s_wait_asynccnt)
#    define USE_ASYNC_LDS 1
#  endif
#endif
#ifndef USE_ASYNC_LDS
#  define USE_ASYNC_LDS 0
#endif

#if USE_ASYNC_LDS
typedef int i4v __attribute__((vector_size(16)));
typedef __attribute__((address_space(1))) i4v gi4;   /* global int4 */
typedef __attribute__((address_space(3))) i4v li4;   /* LDS int4    */
#endif

__device__ __forceinline__ unsigned short bf16bits(float x){
  unsigned u = __float_as_uint(x);
  unsigned r = u + 0x7FFFu + ((u >> 16) & 1u);
  return (unsigned short)(r >> 16);
}
__device__ __forceinline__ float bf16f(unsigned short b){
  return __uint_as_float(((unsigned)b) << 16);
}
__device__ __forceinline__ float softplusf_(float x){
  return (x > 20.f) ? x : log1pf(__expf(x));
}
__device__ __forceinline__ float sigmoidf_(float x){
  return 1.f / (1.f + __expf(-x));
}

/* ---------------- zero fused + stats (deterministic init each call) -------- */
__global__ void k_zero(float* __restrict__ fused, float* __restrict__ stats){
  unsigned i = blockIdx.x * blockDim.x + threadIdx.x;
  f4 z = {0.f, 0.f, 0.f, 0.f};
  if (i < NELEMT/4) ((f4*)fused)[i] = z;
  if (i < 256) stats[i] = 0.f;
}

/* ---------------- STSE conv 1->16, 3x3, fused BN-stat accumulation -------- */
__global__ __launch_bounds__(256)
void k_stse(const float* __restrict__ xin, const float* __restrict__ w,
            float* __restrict__ u, float* __restrict__ slot){
  __shared__ float wsh[144];
  __shared__ float ls[32];
  int tid = threadIdx.x;
  if (tid < 144) wsh[tid] = w[tid];
  if (tid < 32)  ls[tid] = 0.f;
  __syncthreads();
  int y = blockIdx.x, tb = blockIdx.y, x = tid;
  int t = tb >> 2, b = tb & 3;
  const float* xp = xin + (size_t)(b*TT + t) * HWIMG;
  float v[9];
  #pragma unroll
  for (int dy = 0; dy < 3; ++dy)
    #pragma unroll
    for (int dx = 0; dx < 3; ++dx){
      int yy = y + dy - 1, xx = x + dx - 1;
      v[dy*3+dx] = (yy >= 0 && yy < HH && xx >= 0 && xx < WWI) ? xp[yy*WWI + xx] : 0.f;
    }
  unsigned base = ((unsigned)tb*HWIMG + (unsigned)y*WWI + (unsigned)x) * CHN;
  #pragma unroll
  for (int c = 0; c < CHN; ++c){
    float acc = 0.f;
    #pragma unroll
    for (int k = 0; k < 9; ++k) acc = fmaf(wsh[c*9 + k], v[k], acc);
    u[base + c] = acc;
    atomicAdd(&ls[c], acc);
    atomicAdd(&ls[16 + c], acc*acc);
  }
  __syncthreads();
  if (tid < 32) atomicAdd(&slot[tid], ls[tid]);
}

/* ---------------- LIF scan over T for the STSE stage ---------------------- */
__global__ __launch_bounds__(256)
void k_lif0(const float* __restrict__ u, unsigned short* __restrict__ s,
            const float* __restrict__ slot, const float* __restrict__ bng,
            const float* __restrict__ bnb, const float* __restrict__ tau_raw,
            const float* __restrict__ vth_raw){
  unsigned p = blockIdx.x * 256u + threadIdx.x;   /* < BB*HWIMG */
  int b = p >> 16; unsigned pix = p & 65535u;
  const float invN = 1.f / (float)NPIXT;
  float scale[CHN], bias[CHN];
  #pragma unroll
  for (int c = 0; c < CHN; ++c){
    float m = slot[c] * invN;
    float var = fmaxf(slot[16 + c] * invN - m*m, 0.f);
    float sc = bng[c] * rsqrtf(var + 1e-5f);
    scale[c] = sc; bias[c] = bnb[c] - m * sc;
  }
  float tau = softplusf_(tau_raw[0]) + 1.f;
  float decay = __expf(-1.f / tau);
  float vth = softplusf_(vth_raw[0]) + 0.5f;
  float v[CHN];
  #pragma unroll
  for (int c = 0; c < CHN; ++c) v[c] = 0.f;
  for (int t = 0; t < TT; ++t){
    unsigned base = (((unsigned)(t*BB + b)) * HWIMG + pix) * CHN;
    if (t + 1 < TT)
      __builtin_prefetch(u + (((unsigned)((t+1)*BB + b)) * HWIMG + pix) * CHN, 0, 3);
    #pragma unroll
    for (int c = 0; c < CHN; ++c){
      float um = fmaf(u[base + c], scale[c], bias[c]);
      v[c] = fmaf(v[c], decay, um);
      float sp = (v[c] > vth) ? 1.f : 0.f;
      v[c] -= v[c] * sp;
      s[base + c] = bf16bits(sp);
    }
  }
}

/* ---------------- GRU + LayerNorm + FiLM head (tiny, one workgroup) ------- */
__global__ __launch_bounds__(256)
void k_gru(const float* __restrict__ ctx,
           const float* __restrict__ Wih, const float* __restrict__ Whh,
           const float* __restrict__ bih, const float* __restrict__ bhh,
           const float* __restrict__ lng, const float* __restrict__ lnb,
           const float* __restrict__ projw, const float* __restrict__ projb,
           const float* __restrict__ gw, const float* __restrict__ gb,
           const float* __restrict__ bw, const float* __restrict__ bb2,
           const float* __restrict__ tw, const float* __restrict__ tb2,
           const float* __restrict__ vw, const float* __restrict__ vb2,
           const float* __restrict__ aw, const float* __restrict__ ab2,
           float* __restrict__ fgamma, float* __restrict__ fbeta,
           float* __restrict__ fdecay, float* __restrict__ fvth,
           float* __restrict__ falpha){
  __shared__ float hs[4][64];
  __shared__ float ys[4][64];
  __shared__ float als[4][3];
  int tid = threadIdx.x;
  int b = tid >> 6, j = tid & 63;
  hs[b][j] = 0.f;
  __syncthreads();
  for (int t = 0; t < TT; ++t){
    float x0 = ctx[(b*TT + t)*4 + 0], x1 = ctx[(b*TT + t)*4 + 1];
    float x2 = ctx[(b*TT + t)*4 + 2], x3 = ctx[(b*TT + t)*4 + 3];
    float gir = bih[j]      + Wih[j*4+0]*x0 + Wih[j*4+1]*x1 + Wih[j*4+2]*x2 + Wih[j*4+3]*x3;
    float giz = bih[64+j]   + Wih[(64+j)*4+0]*x0 + Wih[(64+j)*4+1]*x1 + Wih[(64+j)*4+2]*x2 + Wih[(64+j)*4+3]*x3;
    float gin = bih[128+j]  + Wih[(128+j)*4+0]*x0 + Wih[(128+j)*4+1]*x1 + Wih[(128+j)*4+2]*x2 + Wih[(128+j)*4+3]*x3;
    float ghr = bhh[j], ghz = bhh[64+j], ghn = bhh[128+j];
    for (int k = 0; k < 64; ++k){
      float hk = hs[b][k];
      ghr = fmaf(Whh[j*64+k], hk, ghr);
      ghz = fmaf(Whh[(64+j)*64+k], hk, ghz);
      ghn = fmaf(Whh[(128+j)*64+k], hk, ghn);
    }
    float r  = sigmoidf_(gir + ghr);
    float zz = sigmoidf_(giz + ghz);
    float n  = tanhf(gin + r * ghn);
    float hn = (1.f - zz)*n + zz*hs[b][j];
    __syncthreads();
    hs[b][j] = hn;
    __syncthreads();
  }
  float msum = 0.f;
  for (int k = 0; k < 64; ++k) msum += hs[b][k];
  float mean = msum * (1.f/64.f);
  float vs = 0.f;
  for (int k = 0; k < 64; ++k){ float d = hs[b][k]-mean; vs = fmaf(d, d, vs); }
  float y = (hs[b][j]-mean)*rsqrtf(vs*(1.f/64.f)+1e-5f)*lng[j] + lnb[j];
  ys[b][j] = y;
  __syncthreads();
  float zj = projb[j];
  for (int k = 0; k < 64; ++k) zj = fmaf(projw[j*64+k], ys[b][k], zj);
  zj = fmaxf(zj, 0.f);
  __syncthreads();
  hs[b][j] = zj;           /* hs now holds z */
  __syncthreads();
  if (tid < 192){
    int b2 = tid / 48, o = tid % 48;
    float dg = gb[o], db = bb2[o];
    for (int k = 0; k < 64; ++k){
      float zk = hs[b2][k];
      dg = fmaf(gw[o*64+k], zk, dg);
      db = fmaf(bw[o*64+k], zk, db);
    }
    fgamma[b2*48+o] = 1.f + 0.1f*tanhf(dg);
    fbeta [b2*48+o] = 0.1f*tanhf(db);
  }
  if (tid < 12){
    int b2 = tid / 3, i = tid % 3;
    float dt = tb2[i], dv = vb2[i], da = ab2[i];
    for (int k = 0; k < 64; ++k){
      float zk = hs[b2][k];
      dt = fmaf(tw[i*64+k], zk, dt);
      dv = fmaf(vw[i*64+k], zk, dv);
      da = fmaf(aw[i*64+k], zk, da);
    }
    float tau = softplusf_(dt) + 1.f;
    fdecay[b2*3+i] = __expf(-1.f/tau);
    fvth  [b2*3+i] = softplusf_(dv) + 0.5f;
    als[b2][i] = da;
  }
  __syncthreads();
  if (tid < 12){
    int b2 = tid / 3, i = tid % 3;
    float mx = fmaxf(als[b2][0], fmaxf(als[b2][1], als[b2][2]));
    float e0 = __expf(als[b2][0]-mx), e1 = __expf(als[b2][1]-mx), e2 = __expf(als[b2][2]-mx);
    falpha[b2*3+i] = __expf(als[b2][i]-mx) / (e0+e1+e2);
  }
}

/* ---------------- 3x3 16->16 conv via V_WMMA_F32_16X16X32_BF16 ------------ */
__device__ __forceinline__ void pix_copy_sync(unsigned short* dst, const unsigned short* src){
  ((u4*)dst)[0] = ((const u4*)src)[0];
  ((u4*)dst)[1] = ((const u4*)src)[1];
}

/* bf16 pixel: async global->LDS when the toolchain has the gfx1250 builtin */
__device__ __forceinline__ void stage_pix(unsigned short* dst, const unsigned short* src){
#if USE_ASYNC_LDS
  gi4* g = (gi4*)(unsigned long long)(uintptr_t)src;
  li4* l = (li4*)(unsigned)(uintptr_t)dst;
  __builtin_amdgcn_global_load_async_to_lds_b128(g, l, 0, 0);
  __builtin_amdgcn_global_load_async_to_lds_b128(g, l, 16, 0);  /* offset applies to both sides */
#else
  pix_copy_sync(dst, src);
#endif
}
/* f32 pixel (post conv): convert to bf16 in the copy */
__device__ __forceinline__ void stage_pix(unsigned short* dst, const float* src){
  #pragma unroll
  for (int c = 0; c < CHN; ++c) dst[c] = bf16bits(src[c]);
}

template<typename TI>
__global__ __launch_bounds__(256)
void k_conv3x3(const TI* __restrict__ sin, const float* __restrict__ w,
               float* __restrict__ uout, float* __restrict__ slot){
  __shared__ __align__(32) unsigned short St[10*130*16];  /* [row][x][cin] NHWC tile + halo */
  __shared__ __align__(32) unsigned short Wl[10*16*16];   /* [tap][cout][cin], tap9 = 0 pad */
  __shared__ float lstat[32];
  int tid = threadIdx.x;
  int xb = blockIdx.x, yb = blockIdx.y, tb = blockIdx.z;

  for (int i = tid; i < 2560; i += 256){
    int tap = i >> 8; int rem = i & 255; int co = rem >> 4; int ci = rem & 15;
    float val = (tap < 9) ? w[(co*16 + ci)*9 + tap] : 0.f;
    Wl[i] = bf16bits(val);
  }
  if (tid < 32) lstat[tid] = 0.f;

  int y0 = yb * 8, x0 = xb * 128;
  for (int cell = tid; cell < 1300; cell += 256){
    int r = cell / 130; int xx = cell - r*130;
    int gy = y0 - 1 + r, gx = x0 - 1 + xx;
    unsigned short* dst = &St[cell * 16];
    if (gy >= 0 && gy < HH && gx >= 0 && gx < WWI){
      stage_pix(dst, sin + ((size_t)tb*HWIMG + (size_t)gy*WWI + gx) * CHN);
    } else {
      u4 z = {0u,0u,0u,0u};
      ((u4*)dst)[0] = z; ((u4*)dst)[1] = z;
    }
  }
#if USE_ASYNC_LDS
  if (sizeof(TI) == 2) __builtin_amdgcn_s_wait_asynccnt(0);
#endif
  __syncthreads();

  int lane = tid & 31, wv = tid >> 5;
  int m = lane & 15, h2 = lane >> 4;

  /* A: weights, contiguous k order (lane half h: tap 2j+h, cin 0..15); held in VGPRs for the whole block */
  v16bf a[5];
  #pragma unroll
  for (int j = 0; j < 5; ++j){
    AU au;
    au.q[0] = *(const u4*)&Wl[(((2*j+h2)*16) + m)*16];
    au.q[1] = *(const u4*)&Wl[(((2*j+h2)*16) + m)*16 + 8];
    a[j] = au.v;
  }

  int yl = wv;   /* each wave owns one output row of the 8-row band */
  float ssum[8] = {0,0,0,0,0,0,0,0}, ssq[8] = {0,0,0,0,0,0,0,0};
  unsigned rowbase = ((unsigned)tb*HWIMG + (unsigned)(y0+yl)*WWI + (unsigned)x0) * CHN;

  for (int ct = 0; ct < 8; ++ct){
    int xl = ct*16 + m;                 /* output pixel (N index) within 128-col band */
    v8f c = {0.f,0.f,0.f,0.f,0.f,0.f,0.f,0.f};
    #pragma unroll
    for (int j = 0; j < 5; ++j){
      int tap = 2*j + h2; tap = (tap > 8) ? 8 : tap;
      int dy = tap / 3, dx = tap - dy*3;
      const v16bf bv = *(const v16bf*)&St[(((yl+dy)*130) + (xl+dx)) * 16];
      c = __builtin_amdgcn_wmma_f32_16x16x32_bf16(false, a[j], false, bv,
                                                  (short)0, c, false, false);
    }
    unsigned ob = rowbase + (unsigned)xl * CHN + 8u*h2;
    f4 lo = {c[0], c[1], c[2], c[3]};
    f4 hi = {c[4], c[5], c[6], c[7]};
    *(f4*)(uout + ob)     = lo;
    *(f4*)(uout + ob + 4) = hi;
    #pragma unroll
    for (int r = 0; r < 8; ++r){ ssum[r] += c[r]; ssq[r] += c[r]*c[r]; }
  }
  #pragma unroll
  for (int r = 0; r < 8; ++r){
    atomicAdd(&lstat[8*h2 + r],      ssum[r]);
    atomicAdd(&lstat[16 + 8*h2 + r], ssq[r]);
  }
  __syncthreads();
  if (tid < 32) atomicAdd(&slot[tid], lstat[tid]);
}

/* ---------------- LIF scan for residual blocks (BN + FiLM + scan) --------- */
__global__ __launch_bounds__(256)
void k_lif_block(const float* __restrict__ u, unsigned short* __restrict__ s,
                 const float* __restrict__ slot, const float* __restrict__ bng,
                 const float* __restrict__ bnb, const float* __restrict__ fgamma,
                 const float* __restrict__ fbeta, const float* __restrict__ fdecay,
                 const float* __restrict__ fvth, int blk){
  unsigned p = blockIdx.x * 256u + threadIdx.x;
  int b = p >> 16; unsigned pix = p & 65535u;
  const float invN = 1.f / (float)NPIXT;
  float scale[CHN], bias[CHN];
  #pragma unroll
  for (int c = 0; c < CHN; ++c){
    float m = slot[c] * invN;
    float var = fmaxf(slot[16 + c] * invN - m*m, 0.f);
    float sc = bng[c] * rsqrtf(var + 1e-5f);
    float g = fgamma[b*48 + blk*16 + c];
    float be = fbeta[b*48 + blk*16 + c];
    scale[c] = sc * g;
    bias[c]  = (bnb[c] - m * sc) * g + be;
  }
  float decay = fdecay[b*3 + blk];
  float vth   = fvth  [b*3 + blk];
  float v[CHN];
  #pragma unroll
  for (int c = 0; c < CHN; ++c) v[c] = 0.f;
  for (int t = 0; t < TT; ++t){
    unsigned base = (((unsigned)(t*BB + b)) * HWIMG + pix) * CHN;
    if (t + 1 < TT){
      unsigned nb = (((unsigned)((t+1)*BB + b)) * HWIMG + pix) * CHN;
      __builtin_prefetch(u + nb, 0, 3);
      __builtin_prefetch(s + nb, 1, 3);
    }
    #pragma unroll
    for (int c = 0; c < CHN; ++c){
      float um = fmaf(u[base + c], scale[c], bias[c]);
      v[c] = fmaf(v[c], decay, um);
      float sp = (v[c] > vth) ? 1.f : 0.f;
      v[c] -= v[c] * sp;
      float snew = bf16f(s[base + c]) + sp;   /* s += spike (exact in bf16) */
      s[base + c] = bf16bits(snew);
    }
  }
}

/* ---------------- fusion 1x1 conv: stats pass then accumulate pass -------- */
__global__ __launch_bounds__(256)
void k_fuse_stats(const unsigned short* __restrict__ s, const float* __restrict__ w,
                  float* __restrict__ slot){
  __shared__ float M[256];
  __shared__ float ls[32];
  int tid = threadIdx.x;
  M[tid] = w[tid];
  if (tid < 32) ls[tid] = 0.f;
  __syncthreads();
  unsigned q = blockIdx.x * 256u + tid;
  float sv[CHN];
  const unsigned short* sp = s + (size_t)q * CHN;
  #pragma unroll
  for (int ci = 0; ci < CHN; ++ci) sv[ci] = bf16f(sp[ci]);
  #pragma unroll
  for (int o = 0; o < CHN; ++o){
    float f = 0.f;
    #pragma unroll
    for (int ci = 0; ci < CHN; ++ci) f = fmaf(M[o*16+ci], sv[ci], f);
    atomicAdd(&ls[o], f);
    atomicAdd(&ls[16+o], f*f);
  }
  __syncthreads();
  if (tid < 32) atomicAdd(&slot[tid], ls[tid]);
}

__global__ __launch_bounds__(256)
void k_fuse_accum(const unsigned short* __restrict__ s, const float* __restrict__ w,
                  const float* __restrict__ slot, const float* __restrict__ bng,
                  const float* __restrict__ bnb, const float* __restrict__ falpha,
                  int blk, float* __restrict__ fused){
  __shared__ float M[256];
  int tid = threadIdx.x;
  M[tid] = w[tid];
  __syncthreads();
  unsigned q = blockIdx.x * 256u + tid;
  int tb = q >> 16; int b = tb & 3;
  float alpha = falpha[b*3 + blk];
  const float invN = 1.f / (float)NPIXT;
  float sv[CHN];
  const unsigned short* sp = s + (size_t)q * CHN;
  #pragma unroll
  for (int ci = 0; ci < CHN; ++ci) sv[ci] = bf16f(sp[ci]);
  float* fp = fused + (size_t)q * CHN;
  #pragma unroll
  for (int o = 0; o < CHN; ++o){
    float f = 0.f;
    #pragma unroll
    for (int ci = 0; ci < CHN; ++ci) f = fmaf(M[o*16+ci], sv[ci], f);
    float m = slot[o] * invN;
    float var = fmaxf(slot[16+o] * invN - m*m, 0.f);
    float sc = bng[o] * rsqrtf(var + 1e-5f);
    float fb = (f - m) * sc + bnb[o];
    fp[o] += fb * alpha;
  }
}

/* ---------------- post-BN, temporal mean, 1x1 head, sigmoid --------------- */
__global__ __launch_bounds__(256)
void k_mean_head(const float* __restrict__ u, const float* __restrict__ slot,
                 const float* __restrict__ bng, const float* __restrict__ bnb,
                 const float* __restrict__ headw, const float* __restrict__ headb,
                 float* __restrict__ out){
  unsigned p = blockIdx.x * 256u + threadIdx.x;
  int b = p >> 16; unsigned pix = p & 65535u;
  const float invN = 1.f / (float)NPIXT;
  float acc[CHN];
  #pragma unroll
  for (int c = 0; c < CHN; ++c) acc[c] = 0.f;
  for (int t = 0; t < TT; ++t){
    unsigned base = (((unsigned)(t*BB + b)) * HWIMG + pix) * CHN;
    if (t + 1 < TT)
      __builtin_prefetch(u + (((unsigned)((t+1)*BB + b)) * HWIMG + pix) * CHN, 0, 3);
    #pragma unroll
    for (int c = 0; c < CHN; ++c) acc[c] += u[base + c];
  }
  float val = headb[0];
  #pragma unroll
  for (int c = 0; c < CHN; ++c){
    float m = slot[c] * invN;
    float var = fmaxf(slot[16+c] * invN - m*m, 0.f);
    float sc = bng[c] * rsqrtf(var + 1e-5f);
    float feat = (acc[c] * (1.f/12.f) - m) * sc + bnb[c];
    val = fmaf(headw[c], feat, val);
  }
  out[p] = sigmoidf_(val);
}

/* ========================================================================== */
extern "C" void kernel_launch(void* const* d_in, const int* in_sizes, int n_in,
                              void* d_out, int out_size, void* d_ws, size_t ws_size,
                              hipStream_t stream){
  (void)in_sizes; (void)n_in; (void)out_size; (void)ws_size;
  /* setup_inputs() order: x_sic, context, then params dict flattened:
     2:stse_w 3:stse_b 4:stse_bng 5:stse_bnb 6:stse_tau_raw 7:stse_vth_raw
     8:gru_wih 9:gru_whh 10:gru_bih 11:gru_bhh 12:ln_g 13:ln_b 14:proj_w 15:proj_b
     16:fc_gamma_w 17:fc_gamma_b 18:fc_beta_w 19:fc_beta_b 20:fc_tau_w 21:fc_tau_b
     22:fc_vth_w 23:fc_vth_b 24:fc_alpha_w 25:fc_alpha_b
     26..34: blocks[3] x (w,bng,bnb)   35..43: fusion_proj[3] x (w,bng,bnb)
     44:post_w 45:post_bng 46:post_bnb 47:head_w 48:head_b */
  const float* x_sic    = (const float*)d_in[0];
  const float* ctx      = (const float*)d_in[1];
  const float* stse_w   = (const float*)d_in[2];
  /* d_in[3] stse_b: cancelled exactly by BatchNorm */
  const float* stse_bng = (const float*)d_in[4];
  const float* stse_bnb = (const float*)d_in[5];
  const float* tau_raw  = (const float*)d_in[6];
  const float* vth_raw  = (const float*)d_in[7];
  const float* gru_wih  = (const float*)d_in[8];
  const float* gru_whh  = (const float*)d_in[9];
  const float* gru_bih  = (const float*)d_in[10];
  const float* gru_bhh  = (const float*)d_in[11];
  const float* ln_g     = (const float*)d_in[12];
  const float* ln_b     = (const float*)d_in[13];
  const float* proj_w   = (const float*)d_in[14];
  const float* proj_b   = (const float*)d_in[15];
  const float* fcg_w    = (const float*)d_in[16];
  const float* fcg_b    = (const float*)d_in[17];
  const float* fcb_w    = (const float*)d_in[18];
  const float* fcb_b    = (const float*)d_in[19];
  const float* fct_w    = (const float*)d_in[20];
  const float* fct_b    = (const float*)d_in[21];
  const float* fcv_w    = (const float*)d_in[22];
  const float* fcv_b    = (const float*)d_in[23];
  const float* fca_w    = (const float*)d_in[24];
  const float* fca_b    = (const float*)d_in[25];
  const float* blk_w[3]   = {(const float*)d_in[26], (const float*)d_in[29], (const float*)d_in[32]};
  const float* blk_bng[3] = {(const float*)d_in[27], (const float*)d_in[30], (const float*)d_in[33]};
  const float* blk_bnb[3] = {(const float*)d_in[28], (const float*)d_in[31], (const float*)d_in[34]};
  const float* fus_w[3]   = {(const float*)d_in[35], (const float*)d_in[38], (const float*)d_in[41]};
  const float* fus_bng[3] = {(const float*)d_in[36], (const float*)d_in[39], (const float*)d_in[42]};
  const float* fus_bnb[3] = {(const float*)d_in[37], (const float*)d_in[40], (const float*)d_in[43]};
  const float* post_w   = (const float*)d_in[44];
  const float* post_bng = (const float*)d_in[45];
  const float* post_bnb = (const float*)d_in[46];
  const float* head_w   = (const float*)d_in[47];
  const float* head_b   = (const float*)d_in[48];
  float* out = (float*)d_out;

  char* ws = (char*)d_ws;
  float* u      = (float*)ws;                            /* 201.3 MB f32  */
  float* fused  = u + NELEMT;                            /* 201.3 MB f32  */
  unsigned short* s = (unsigned short*)(fused + NELEMT); /* 100.7 MB bf16 */
  float* stats  = (float*)(s + NELEMT);                  /* 8 slots x 32  */
  float* fgamma = stats + 256;                           /* 192 */
  float* fbeta  = fgamma + 192;                          /* 192 */
  float* fdecay = fbeta + 192;                           /* 12  */
  float* fvth   = fdecay + 12;                           /* 12  */
  float* falpha = fvth + 12;                             /* 12  */

  k_zero<<<NELEMT/4/256, 256, 0, stream>>>(fused, stats);
  k_stse<<<dim3(HH, TBN), 256, 0, stream>>>(x_sic, stse_w, u, stats + 0*32);
  k_lif0<<<BB*HWIMG/256, 256, 0, stream>>>(u, s, stats + 0*32, stse_bng, stse_bnb, tau_raw, vth_raw);
  k_gru<<<1, 256, 0, stream>>>(ctx, gru_wih, gru_whh, gru_bih, gru_bhh, ln_g, ln_b,
                               proj_w, proj_b, fcg_w, fcg_b, fcb_w, fcb_b,
                               fct_w, fct_b, fcv_w, fcv_b, fca_w, fca_b,
                               fgamma, fbeta, fdecay, fvth, falpha);
  for (int i = 0; i < 3; ++i){
    k_conv3x3<unsigned short><<<dim3(2, 32, TBN), 256, 0, stream>>>(s, blk_w[i], u, stats + (1+i)*32);
    k_lif_block<<<BB*HWIMG/256, 256, 0, stream>>>(u, s, stats + (1+i)*32, blk_bng[i], blk_bnb[i],
                                                  fgamma, fbeta, fdecay, fvth, i);
    k_fuse_stats<<<NPIXT/256, 256, 0, stream>>>(s, fus_w[i], stats + (4+i)*32);
    k_fuse_accum<<<NPIXT/256, 256, 0, stream>>>(s, fus_w[i], stats + (4+i)*32,
                                                fus_bng[i], fus_bnb[i], falpha, i, fused);
  }
  k_conv3x3<float><<<dim3(2, 32, TBN), 256, 0, stream>>>(fused, post_w, u, stats + 7*32);
  k_mean_head<<<BB*HWIMG/256, 256, 0, stream>>>(u, stats + 7*32, post_bng, post_bnb,
                                                head_w, head_b, out);
}
